// SimpleAttention_49709951484568
// MI455X (gfx1250) — matrix operand order repaired and checked by reading, not compiled, not run on hardware
//
#include <hip/hip_runtime.h>
#include <cstddef>
#include <cstdint>

// ---------------------------------------------------------------------------
// SimpleAttention for MI455X (gfx1250): bf16 WMMA pipeline.
//   B=4, S=4096, HIDDEN=1024, QKV=1024, H=16, D=64, M = B*S = 16384
// Roofline: 107 GFLOP total (96% in the QKV projections) vs ~160 MB HBM
// traffic -> compute-bound at bf16 WMMA rates; everything runs on
// v_wmma_f32_16x16x32_bf16 except the tiny K^T V reduction (2% of FLOPs).
// ---------------------------------------------------------------------------

typedef __bf16 v16bf __attribute__((ext_vector_type(16)));
typedef __bf16 v8bf  __attribute__((ext_vector_type(8)));
typedef __bf16 v4bf  __attribute__((ext_vector_type(4)));
typedef float  v8f   __attribute__((ext_vector_type(8)));
typedef float  v4f   __attribute__((ext_vector_type(4)));

#define NB   4
#define NS   4096
#define NH   16
#define ND   64
#define HID  1024
#define NM   (NB * NS)          // 16384 rows
#define NQKV 1024

// workspace layout (bytes)
#define XB_BYTES   ((size_t)NM * HID * 2)            // x as bf16        32 MB
#define QKV_BYTES  ((size_t)NB * NH * NS * ND * 2)   // each of Q/K/V    32 MB
#define WT_BYTES   ((size_t)3 * NQKV * HID * 2)      // W^T bf16          6 MB

static __device__ __forceinline__ v16bf mkfrag(const __bf16* p0, const __bf16* p1) {
  v8bf lo = *reinterpret_cast<const v8bf*>(p0);
  v8bf hi = *reinterpret_cast<const v8bf*>(p1);
  return __builtin_shufflevector(lo, hi, 0,1,2,3,4,5,6,7,8,9,10,11,12,13,14,15);
}

// --------------------------- prep: x fp32 -> bf16 ---------------------------
__global__ __launch_bounds__(256) void xprep(const float* __restrict__ x,
                                             __bf16* __restrict__ xb) {
  int i = blockIdx.x * 256 + threadIdx.x;          // over n/4 elements
  v4f xv = reinterpret_cast<const v4f*>(x)[i];
  v4bf o;
  o[0] = (__bf16)xv[0]; o[1] = (__bf16)xv[1];
  o[2] = (__bf16)xv[2]; o[3] = (__bf16)xv[3];
  reinterpret_cast<v4bf*>(xb)[i] = o;
}

// ----------------- prep: W [K][N] fp32 -> W^T [3*N][K] bf16 -----------------
__global__ __launch_bounds__(256) void wt_prep(const float* __restrict__ Wq,
                                               const float* __restrict__ Wk,
                                               const float* __restrict__ Wv,
                                               __bf16* __restrict__ WT) {
  __shared__ float t[32][33];
  const int m = blockIdx.z;
  const float* W = (m == 0) ? Wq : ((m == 1) ? Wk : Wv);
  const int k0 = blockIdx.x * 32, n0 = blockIdx.y * 32;
  const int tx = threadIdx.x & 31, ty = threadIdx.x >> 5;   // ty: 0..7
#pragma unroll
  for (int i = 0; i < 4; ++i) {
    int k = k0 + ty + i * 8;
    t[ty + i * 8][tx] = W[(size_t)k * NQKV + n0 + tx];      // coalesced read
  }
  __syncthreads();
#pragma unroll
  for (int i = 0; i < 4; ++i) {
    int n = n0 + ty + i * 8;
    WT[((size_t)m * NQKV + n) * HID + k0 + tx] = (__bf16)t[tx][ty + i * 8];
  }
}

// --------------------- QKV projection GEMM (bf16 WMMA) ----------------------
// C[16384, 3072] = Xb[16384,1024] @ W[1024,3072] (+bias, *mask)
// WG tile 128x256, 8 waves (2m x 4n), wave tile 64x64 = 16 accumulators,
// K-step 32. Fragments loaded straight from global in WMMA-native layout
// (2 x b128 each); intra-WGP operand reuse is served by the WGP read cache,
// keeping the hot loop barrier-free.  16 b128 loads : 16 WMMAs per K-step.
__global__ __launch_bounds__(256) void qkv_gemm(const __bf16* __restrict__ xb,
                                                const __bf16* __restrict__ wt,
                                                const float* __restrict__ bq,
                                                const float* __restrict__ bk,
                                                const float* __restrict__ bv,
                                                const float* __restrict__ maskp,
                                                __bf16* __restrict__ Qb,
                                                __bf16* __restrict__ Kb,
                                                __bf16* __restrict__ Vb) {
  const int wid  = threadIdx.x >> 5;
  const int lane = threadIdx.x & 31;
  const int half = lane >> 4;          // 0: lanes 0-15, 1: lanes 16-31
  const int l    = lane & 15;

  const int m_wave = blockIdx.x * 128 + (wid & 1) * 64;
  const int n_wave = blockIdx.y * 256 + (wid >> 1) * 64;

  v8f acc[4][4];
  const v8f vz = {0.f, 0.f, 0.f, 0.f, 0.f, 0.f, 0.f, 0.f};
#pragma unroll
  for (int mi = 0; mi < 4; ++mi)
#pragma unroll
    for (int nj = 0; nj < 4; ++nj) acc[mi][nj] = vz;

  for (int k0 = 0; k0 < HID; k0 += 32) {
    v16bf a[4], bb[4];
#pragma unroll
    for (int mi = 0; mi < 4; ++mi) {
      const __bf16* rp = xb + (size_t)(m_wave + mi * 16 + l) * HID + k0 + half * 8;
      a[mi] = mkfrag(rp, rp + 16);
      __builtin_prefetch(rp + 32, 0, 1);                    // next K-step (speculative)
    }
#pragma unroll
    for (int nj = 0; nj < 4; ++nj) {
      const __bf16* rp = wt + (size_t)(n_wave + nj * 16 + l) * HID + k0 + half * 8;
      bb[nj] = mkfrag(rp, rp + 16);
    }
#pragma unroll
    for (int mi = 0; mi < 4; ++mi)
#pragma unroll
      for (int nj = 0; nj < 4; ++nj)
        acc[mi][nj] = __builtin_amdgcn_wmma_f32_16x16x32_bf16(
            false, a[mi], false, bb[nj], (short)0, acc[mi][nj], false, false);
  }

  // epilogue: +bias, *mask, store bf16 as [B][H][S][64]
  const int mat = blockIdx.y >> 2;     // 256-col blocks, 4 per matrix
  const float* bias = (mat == 0) ? bq : ((mat == 1) ? bk : bv);
  __bf16* outp      = (mat == 0) ? Qb : ((mat == 1) ? Kb : Vb);

  float mrow[4][8];
#pragma unroll
  for (int mi = 0; mi < 4; ++mi)
#pragma unroll
    for (int r = 0; r < 8; ++r)
      mrow[mi][r] = maskp[m_wave + mi * 16 + half * 8 + r];

#pragma unroll
  for (int mi = 0; mi < 4; ++mi)
#pragma unroll
    for (int nj = 0; nj < 4; ++nj) {
      const int ncol = n_wave + nj * 16 + l;
      const int c = ncol & (NQKV - 1);
      const int h = c >> 6, d = c & 63;
      const float bvv = bias[c];
#pragma unroll
      for (int r = 0; r < 8; ++r) {
        const int M = m_wave + mi * 16 + half * 8 + r;
        const int b = M >> 12, s = M & (NS - 1);
        const float v = (acc[mi][nj][r] + bvv) * mrow[mi][r];
        outp[(((size_t)(b * NH + h) * NS + s) * ND) + d] = (__bf16)v;
      }
    }
}

// ------------------- kv = K^T V per (b,h)  (LDS-staged) ---------------------
// Staging uses gfx1250 async DMA: global_load_async_to_lds_b128 (ASYNCcnt) +
// s_wait_asynccnt, bypassing VGPRs.  Writes kv^T (bf16, norm applied) so the
// next GEMM's B operand is contiguous.
__global__ __launch_bounds__(256) void kv_kernel(const __bf16* __restrict__ Kb,
                                                 const __bf16* __restrict__ Vb,
                                                 const float* __restrict__ maskp,
                                                 __bf16* __restrict__ kvT) {
  __shared__ __bf16 KcB[32][64];
  __shared__ __bf16 VcB[32][64];
  __shared__ float red[256];

  const int bh = blockIdx.x;
  const int b  = bh >> 4;
  const int tid = threadIdx.x;

  // norm[b] = 1/sqrt(sum(mask[b]) * HEAD_DIM)
  float msum = 0.f;
  for (int t = tid; t < NS; t += 256) msum += maskp[b * NS + t];
  red[tid] = msum;
  __syncthreads();
  for (int off = 128; off > 0; off >>= 1) {
    if (tid < off) red[tid] += red[tid + off];
    __syncthreads();
  }
  const float norm = rsqrtf(red[0] * (float)ND);

  const int td = tid & 15, te = tid >> 4;
  float acc[4][4] = {};
  const size_t base = (size_t)bh * NS * ND;

  // per-thread LDS destinations (ISA: LDS offset = low 32 bits of generic addr)
  const unsigned ldsK = (unsigned)(uintptr_t)(&KcB[0][0]) + (unsigned)tid * 16u;
  const unsigned ldsV = (unsigned)(uintptr_t)(&VcB[0][0]) + (unsigned)tid * 16u;

  for (int s0 = 0; s0 < NS; s0 += 32) {
    __syncthreads();   // protect LDS tile reuse from previous iteration
    {
      const __bf16* gk = Kb + base + (size_t)s0 * ND + (size_t)tid * 8;
      const __bf16* gv = Vb + base + (size_t)s0 * ND + (size_t)tid * 8;
      asm volatile("global_load_async_to_lds_b128 %0, %1, off"
                   :: "v"(ldsK), "v"(gk) : "memory");
      asm volatile("global_load_async_to_lds_b128 %0, %1, off"
                   :: "v"(ldsV), "v"(gv) : "memory");
      asm volatile("s_wait_asynccnt 0x0" ::: "memory");
    }
    __syncthreads();
#pragma unroll 4
    for (int ss = 0; ss < 32; ++ss) {
      v4bf k4 = *reinterpret_cast<const v4bf*>(&KcB[ss][td * 4]);
      v4bf v4_ = *reinterpret_cast<const v4bf*>(&VcB[ss][te * 4]);
      float kd[4], vd[4];
#pragma unroll
      for (int i = 0; i < 4; ++i) { kd[i] = (float)k4[i]; vd[i] = (float)v4_[i]; }
#pragma unroll
      for (int i = 0; i < 4; ++i)
#pragma unroll
        for (int j = 0; j < 4; ++j) acc[i][j] += kd[i] * vd[j];
    }
  }

#pragma unroll
  for (int i = 0; i < 4; ++i)
#pragma unroll
    for (int j = 0; j < 4; ++j) {
      const int d = td * 4 + i, e = te * 4 + j;
      kvT[((size_t)bh * ND + e) * ND + d] = (__bf16)(acc[i][j] * norm);
    }
}

// ---------------------- out = Q @ kv  (bf16 WMMA) ---------------------------
// per (b,h): M=4096, N=64, K=64.  8 waves x 64 rows = 512 rows per WG.
__global__ __launch_bounds__(256) void out_kernel(const __bf16* __restrict__ Qb,
                                                  const __bf16* __restrict__ kvT,
                                                  float* __restrict__ out) {
  const int bh = blockIdx.y;
  const int b = bh >> 4, h = bh & 15;
  const int wid  = threadIdx.x >> 5;
  const int lane = threadIdx.x & 31;
  const int half = lane >> 4, l = lane & 15;
  const int s_wave = blockIdx.x * 512 + wid * 64;

  v8f acc[4][4];
  const v8f vz = {0.f, 0.f, 0.f, 0.f, 0.f, 0.f, 0.f, 0.f};
#pragma unroll
  for (int mi = 0; mi < 4; ++mi)
#pragma unroll
    for (int nj = 0; nj < 4; ++nj) acc[mi][nj] = vz;

#pragma unroll
  for (int k0 = 0; k0 < ND; k0 += 32) {
    v16bf a[4], bb[4];
#pragma unroll
    for (int mi = 0; mi < 4; ++mi) {
      const __bf16* rp = Qb + ((size_t)bh * NS + s_wave + mi * 16 + l) * ND + k0 + half * 8;
      a[mi] = mkfrag(rp, rp + 16);
    }
#pragma unroll
    for (int nj = 0; nj < 4; ++nj) {
      const __bf16* rp = kvT + ((size_t)bh * ND + nj * 16 + l) * ND + k0 + half * 8;
      bb[nj] = mkfrag(rp, rp + 16);
    }
#pragma unroll
    for (int mi = 0; mi < 4; ++mi)
#pragma unroll
      for (int nj = 0; nj < 4; ++nj)
        acc[mi][nj] = __builtin_amdgcn_wmma_f32_16x16x32_bf16(
            false, a[mi], false, bb[nj], (short)0, acc[mi][nj], false, false);
  }

#pragma unroll
  for (int mi = 0; mi < 4; ++mi)
#pragma unroll
    for (int nj = 0; nj < 4; ++nj) {
      const int e = nj * 16 + l;
#pragma unroll
      for (int r = 0; r < 8; ++r) {
        const int s = s_wave + mi * 16 + half * 8 + r;
        out[((size_t)(b * NS + s)) * NQKV + h * ND + e] = acc[mi][nj][r];
      }
    }
}

// ---------------------------------------------------------------------------
extern "C" void kernel_launch(void* const* d_in, const int* in_sizes, int n_in,
                              void* d_out, int out_size, void* d_ws, size_t ws_size,
                              hipStream_t stream) {
  (void)in_sizes; (void)n_in; (void)out_size; (void)ws_size;
  const float* x    = (const float*)d_in[0];
  const float* mask = (const float*)d_in[1];
  const float* Wq   = (const float*)d_in[2];
  const float* bq   = (const float*)d_in[3];
  const float* Wk   = (const float*)d_in[4];
  const float* bk   = (const float*)d_in[5];
  const float* Wv   = (const float*)d_in[6];
  const float* bv   = (const float*)d_in[7];
  float* out = (float*)d_out;

  char* ws = (char*)d_ws;
  __bf16* xb  = (__bf16*)(ws);
  __bf16* Qb  = (__bf16*)(ws + XB_BYTES);
  __bf16* Kb  = (__bf16*)(ws + XB_BYTES + QKV_BYTES);
  __bf16* Vb  = (__bf16*)(ws + XB_BYTES + 2 * QKV_BYTES);
  __bf16* WT  = (__bf16*)(ws + XB_BYTES + 3 * QKV_BYTES);
  __bf16* kvT = (__bf16*)(ws + XB_BYTES + 3 * QKV_BYTES + WT_BYTES);

  xprep<<<dim3((NM * HID) / 4 / 256), 256, 0, stream>>>(x, xb);
  wt_prep<<<dim3(HID / 32, NQKV / 32, 3), 256, 0, stream>>>(Wq, Wk, Wv, WT);
  qkv_gemm<<<dim3(NM / 128, (3 * NQKV) / 256), 256, 0, stream>>>(
      xb, WT, bq, bk, bv, mask, Qb, Kb, Vb);
  kv_kernel<<<dim3(NB * NH), 256, 0, stream>>>(Kb, Vb, mask, kvT);
  out_kernel<<<dim3(NS / 512, NB * NH), 256, 0, stream>>>(Qb, kvT, out);
}
